// CombinedAttnProcessor_double_41412074668261
// MI455X (gfx1250) — compile-verified, matrix operand order checked by reading, not run on hardware
//
#include <hip/hip_runtime.h>
#include <cstdint>
#include <cstddef>

// ---------------------------------------------------------------------------
// Types for CDNA5 WMMA (wave32): D(f32 16x16) = A(bf16 16x32) * B(bf16 32x16) + C
// ---------------------------------------------------------------------------
typedef __attribute__((ext_vector_type(16))) __bf16 bf16x16;
typedef __attribute__((ext_vector_type(8)))  float  f32x8;
typedef int v4i __attribute__((vector_size(16)));

union Frag16 { bf16x16 f; unsigned int u[8]; unsigned short h[16]; };
union F8     { f32x8 v;   float f[8]; };

__device__ __forceinline__ unsigned short f2bf(float x) {
  unsigned int u = __float_as_uint(x);
  u += 0x7FFFu + ((u >> 16) & 1u);   // round-to-nearest-even
  return (unsigned short)(u >> 16);
}

// ---------------------------------------------------------------------------
// CDNA5 async global->LDS copy (GLOBAL_LOAD_ASYNC_TO_LDS_B128, ASYNCcnt),
// with a VGPR-bounce fallback when the builtins are unavailable.
// ---------------------------------------------------------------------------
#if __has_builtin(__builtin_amdgcn_global_load_async_to_lds_b128) && \
    __has_builtin(__builtin_amdgcn_s_wait_asynccnt)
#define USE_ASYNC_COPY 1
#else
#define USE_ASYNC_COPY 0
#endif

__device__ __forceinline__ void lds_cp_b128(unsigned short* l, const unsigned short* g) {
#if USE_ASYNC_COPY
  __builtin_amdgcn_global_load_async_to_lds_b128(
      (__attribute__((address_space(1))) v4i*)(g),
      (__attribute__((address_space(3))) v4i*)(l), 0, 0);
#else
  *(uint4*)l = *(const uint4*)g;
#endif
}

__device__ __forceinline__ void lds_cp_fence() {
#if USE_ASYNC_COPY
  __builtin_amdgcn_s_wait_asynccnt(0);
#endif
  __syncthreads();
}

// ---------------------------------------------------------------------------
// Elementwise conversion kernels
// ---------------------------------------------------------------------------
__global__ __launch_bounds__(256) void cvt_bf16_kernel(const float* __restrict__ in,
                                                       unsigned short* __restrict__ out,
                                                       long n) {
  long i = (long)blockIdx.x * blockDim.x + threadIdx.x;
  if (i < n) out[i] = f2bf(in[i]);
}

__global__ __launch_bounds__(256) void add_cvt_bf16_kernel(const float* __restrict__ a,
                                                           const float* __restrict__ b,
                                                           unsigned short* __restrict__ out,
                                                           long n) {
  long i = (long)blockIdx.x * blockDim.x + threadIdx.x;
  if (i < n) out[i] = f2bf(a[i] + b[i]);
}

// Tiled transpose + fp32->bf16: out[c][r] = bf16(in[r][c]), R,C multiples of 32.
__global__ __launch_bounds__(256) void cvt_transpose_bf16_kernel(
    const float* __restrict__ in, unsigned short* __restrict__ out, int R, int C) {
  __shared__ unsigned short tile[32][33];
  const int c0 = blockIdx.x * 32;
  const int r0 = blockIdx.y * 32;
  const int tr = threadIdx.x >> 5;
  const int tc = threadIdx.x & 31;
#pragma unroll
  for (int i = 0; i < 4; ++i) {
    const int r = tr + i * 8;
    tile[r][tc] = f2bf(in[(long)(r0 + r) * C + c0 + tc]);
  }
  __syncthreads();
#pragma unroll
  for (int i = 0; i < 4; ++i) {
    const int r = tr + i * 8;
    out[(long)(c0 + r) * R + r0 + tc] = tile[tc][r];
  }
}

// ---------------------------------------------------------------------------
// RMS norm over contiguous groups of 128 floats (one wave per group)
// ---------------------------------------------------------------------------
__global__ __launch_bounds__(256) void rms_kernel(float* __restrict__ x, long ngroups) {
  const long g = (long)blockIdx.x * 8 + (threadIdx.x >> 5);
  const int lane = threadIdx.x & 31;
  if (g >= ngroups) return;
  float* p = x + g * 128;
  float v0 = p[lane], v1 = p[lane + 32], v2 = p[lane + 64], v3 = p[lane + 96];
  float ss = v0 * v0 + v1 * v1 + v2 * v2 + v3 * v3;
#pragma unroll
  for (int m = 1; m < 32; m <<= 1) ss += __shfl_xor(ss, m, 32);
  const float r = rsqrtf(ss * (1.0f / 128.0f) + 1e-5f);
  p[lane] = v0 * r; p[lane + 32] = v1 * r; p[lane + 64] = v2 * r; p[lane + 96] = v3 * r;
}

// ---------------------------------------------------------------------------
// LoRA: down (rank-16 dot per wave) and up (+= per element)
// ---------------------------------------------------------------------------
__global__ __launch_bounds__(256) void lora_dn_kernel(const float* __restrict__ X,   // [rows][3072]
                                                      const float* __restrict__ Dn,  // [3072][16]
                                                      float* __restrict__ T,         // [rows][16]
                                                      int rows) {
  const int g = blockIdx.x * 8 + (threadIdx.x >> 5);
  const int lane = threadIdx.x & 31;
  if (g >= rows * 16) return;
  const int row = g >> 4, rr = g & 15;
  const float* x = X + (long)row * 3072;
  float s = 0.f;
  for (int k = lane; k < 3072; k += 32) s += x[k] * Dn[(long)k * 16 + rr];
#pragma unroll
  for (int m = 1; m < 32; m <<= 1) s += __shfl_xor(s, m, 32);
  if (lane == 0) T[g] = s;
}

__global__ __launch_bounds__(256) void lora_up_add_kernel(const float* __restrict__ T,   // [rows][16]
                                                          const float* __restrict__ Up,  // [16][3072]
                                                          float* __restrict__ Y,         // [rows][3072] (+=)
                                                          long n, float scale) {
  const long i = (long)blockIdx.x * blockDim.x + threadIdx.x;
  if (i >= n) return;
  const int row = (int)(i / 3072), col = (int)(i % 3072);
  float s = 0.f;
#pragma unroll
  for (int k = 0; k < 16; ++k) s += T[row * 16 + k] * Up[(long)k * 3072 + col];
  Y[i] += scale * s;
}

// ---------------------------------------------------------------------------
// bf16 WMMA GEMM: C[M][ldc] = A[M][K](bf16) * BT[N][K](bf16, pre-transposed) + bias
// block tile 128x64, K-slice 32, double-buffered LDS, async global->LDS staging
// ---------------------------------------------------------------------------
__global__ __launch_bounds__(256) void gemm_bf16_kernel(
    const unsigned short* __restrict__ A,
    const unsigned short* __restrict__ BT,
    float* __restrict__ C,
    const float* __restrict__ bias,
    int M, int N, int K, int ldc) {
  __shared__ __align__(16) unsigned short sA[2][128][32];  // [m][k]
  __shared__ __align__(16) unsigned short sB[2][64][32];   // [n][k]
  const int tid  = threadIdx.x;
  const int wave = tid >> 5;
  const int lane = tid & 31;
  const int ln   = lane & 15;
  const int kh   = (lane >> 4) << 3;  // 0 or 8 (A/B layout K-half)
  const int m0 = blockIdx.y * 128;
  const int n0 = blockIdx.x * 64;

  // staging coordinates (all lanes active; row clamped to stay in-bounds)
  const int ar = tid >> 1;               // 0..127
  const int ak = (tid & 1) << 4;         // 0 / 16
  const int arow = (m0 + ar < M) ? (m0 + ar) : (M - 1);
  const int bn = tid >> 2;               // 0..63
  const int bk = (tid & 3) << 3;         // 0,8,16,24

  auto stage = [&](int k0, int buf) {
    const unsigned short* asrc = A + (long)arow * K + k0 + ak;
    lds_cp_b128(&sA[buf][ar][ak], asrc);
    lds_cp_b128(&sA[buf][ar][ak + 8], asrc + 8);
    lds_cp_b128(&sB[buf][bn][bk], BT + (long)(n0 + bn) * K + k0 + bk);
  };

  F8 acc[4];
#pragma unroll
  for (int c = 0; c < 4; ++c)
#pragma unroll
    for (int j = 0; j < 8; ++j) acc[c].f[j] = 0.f;

  stage(0, 0);
  lds_cp_fence();
  int cur = 0;
  for (int k0 = 0; k0 < K; k0 += 32) {
    if (k0 + 32 < K) stage(k0 + 32, cur ^ 1);

    Frag16 af;
    const int mrel = (wave << 4) + ln;
#pragma unroll
    for (int vi = 0; vi < 4; ++vi) {
      af.u[vi]     = *(const unsigned int*)&sA[cur][mrel][kh + 2 * vi];
      af.u[4 + vi] = *(const unsigned int*)&sA[cur][mrel][16 + kh + 2 * vi];
    }
#pragma unroll
    for (int c = 0; c < 4; ++c) {
      Frag16 bfr;
      const int nrel = (c << 4) + ln;
#pragma unroll
      for (int vi = 0; vi < 4; ++vi) {
        bfr.u[vi]     = *(const unsigned int*)&sB[cur][nrel][kh + 2 * vi];
        bfr.u[4 + vi] = *(const unsigned int*)&sB[cur][nrel][16 + kh + 2 * vi];
      }
      acc[c].v = __builtin_amdgcn_wmma_f32_16x16x32_bf16(
          false, af.f, false, bfr.f, (short)0, acc[c].v, false, false);
    }
    lds_cp_fence();
    cur ^= 1;
  }

  const int rowb = m0 + (wave << 4) + ((lane >> 4) << 3);
#pragma unroll
  for (int c = 0; c < 4; ++c) {
    const int col = n0 + (c << 4) + ln;
    const float bv = bias ? bias[col] : 0.0f;
#pragma unroll
    for (int r = 0; r < 8; ++r) {
      const int row = rowb + r;
      if (row < M) C[(long)row * ldc + col] = acc[c].f[r] + bv;
    }
  }
}

// ---------------------------------------------------------------------------
// Flash attention (bf16 WMMA, fp32 accumulate).
// Q,K row-major [row][24*128]; V pre-transposed VT [24*128][klen].
// One wave = 16 q rows; block = 128 q rows of one head; KV streamed in 32-steps.
// q-tiles with q0 >= sbq start at kv = sbkv (tile-aligned block mask).
// ---------------------------------------------------------------------------
__global__ __launch_bounds__(256) void attn_kernel(
    const unsigned short* __restrict__ Q,
    const unsigned short* __restrict__ Kb,
    const unsigned short* __restrict__ VT,
    float* __restrict__ O,
    int qlen, int klen, int sbq, int sbkv, float scale) {
  __shared__ __align__(16) unsigned short sK[32][128];   // [kv][dh]
  __shared__ __align__(16) unsigned short sVt[128][32];  // [dh][kv]
  __shared__ __align__(16) unsigned short sP[8][16][32]; // per-wave P tile
  const int HD = 3072;
  const int tid = threadIdx.x, wave = tid >> 5, lane = tid & 31;
  const int ln = lane & 15, kh = (lane >> 4) << 3;
  const int h = blockIdx.y;
  const int q0 = blockIdx.x << 7;
  const int colh = h << 7;

  // preload Q fragments (16 rows x 128 dh -> 4 A-frags of K=32)
  const int qrow = q0 + (wave << 4) + ln;
  const unsigned short* qp = Q + (long)qrow * HD + colh;
  Frag16 qf[4];
#pragma unroll
  for (int c = 0; c < 4; ++c) {
#pragma unroll
    for (int vi = 0; vi < 4; ++vi) {
      qf[c].u[vi]     = *(const unsigned int*)(qp + (c << 5) + kh + 2 * vi);
      qf[c].u[4 + vi] = *(const unsigned int*)(qp + (c << 5) + 16 + kh + 2 * vi);
    }
  }

  F8 acc[8];
#pragma unroll
  for (int c = 0; c < 8; ++c)
#pragma unroll
    for (int j = 0; j < 8; ++j) acc[c].f[j] = 0.f;
  float mrow[8], lrow[8];
#pragma unroll
  for (int r = 0; r < 8; ++r) { mrow[r] = -1e30f; lrow[r] = 0.f; }

  // staging coordinates
  const int skr = tid >> 3;            // kv row 0..31
  const int skc = (tid & 7) << 4;      // dh col 0..112 step 16
  const int svd = tid >> 1;            // dh row 0..127
  const int svk = (tid & 1) << 4;      // kv col 0 / 16

  const int kvs = (q0 >= sbq) ? sbkv : 0;
  for (int kv = kvs; kv < klen; kv += 32) {
    {  // async stage K (row-major) and V^T (contiguous along kv)
      const unsigned short* kp = Kb + (long)(kv + skr) * HD + colh + skc;
      lds_cp_b128(&sK[skr][skc], kp);
      lds_cp_b128(&sK[skr][skc + 8], kp + 8);
      const unsigned short* vp = VT + (long)(colh + svd) * klen + kv + svk;
      lds_cp_b128(&sVt[svd][svk], vp);
      lds_cp_b128(&sVt[svd][svk + 8], vp + 8);
    }
    lds_cp_fence();

    // S = Q K^T : two 16-col tiles, each accumulated over 4 K-chunks of 32
    F8 s0, s1;
#pragma unroll
    for (int j = 0; j < 8; ++j) { s0.f[j] = 0.f; s1.f[j] = 0.f; }
#pragma unroll
    for (int c = 0; c < 4; ++c) {
      Frag16 kf0, kf1;
#pragma unroll
      for (int vi = 0; vi < 4; ++vi) {
        kf0.u[vi]     = *(const unsigned int*)&sK[ln][(c << 5) + kh + 2 * vi];
        kf0.u[4 + vi] = *(const unsigned int*)&sK[ln][(c << 5) + 16 + kh + 2 * vi];
        kf1.u[vi]     = *(const unsigned int*)&sK[16 + ln][(c << 5) + kh + 2 * vi];
        kf1.u[4 + vi] = *(const unsigned int*)&sK[16 + ln][(c << 5) + 16 + kh + 2 * vi];
      }
      s0.v = __builtin_amdgcn_wmma_f32_16x16x32_bf16(false, qf[c].f, false, kf0.f,
                                                     (short)0, s0.v, false, false);
      s1.v = __builtin_amdgcn_wmma_f32_16x16x32_bf16(false, qf[c].f, false, kf1.f,
                                                     (short)0, s1.v, false, false);
    }

    // online softmax update (per-row stats over the 16-lane column groups)
#pragma unroll
    for (int r = 0; r < 8; ++r) {
      const float a0 = s0.f[r] * scale;
      const float a1 = s1.f[r] * scale;
      float mx = fmaxf(a0, a1);
#pragma unroll
      for (int m = 1; m < 16; m <<= 1) mx = fmaxf(mx, __shfl_xor(mx, m, 32));
      const float nm = fmaxf(mrow[r], mx);
      const float corr = __expf(mrow[r] - nm);
      const float p0 = __expf(a0 - nm);
      const float p1 = __expf(a1 - nm);
      float rs = p0 + p1;
#pragma unroll
      for (int m = 1; m < 16; m <<= 1) rs += __shfl_xor(rs, m, 32);
      lrow[r] = lrow[r] * corr + rs;
      mrow[r] = nm;
#pragma unroll
      for (int c = 0; c < 8; ++c) acc[c].f[r] *= corr;
      const int prow = r + ((lane >> 4) << 3);
      sP[wave][prow][ln]      = f2bf(p0);
      sP[wave][prow][ln + 16] = f2bf(p1);
    }

    // O += P @ V  (P reloaded as A-frag from wave-private LDS, V^T gives B-frags)
    Frag16 pf;
#pragma unroll
    for (int vi = 0; vi < 4; ++vi) {
      pf.u[vi]     = *(const unsigned int*)&sP[wave][ln][kh + 2 * vi];
      pf.u[4 + vi] = *(const unsigned int*)&sP[wave][ln][16 + kh + 2 * vi];
    }
#pragma unroll
    for (int c = 0; c < 8; ++c) {
      Frag16 vf;
      const int n = (c << 4) + ln;
#pragma unroll
      for (int vi = 0; vi < 4; ++vi) {
        vf.u[vi]     = *(const unsigned int*)&sVt[n][kh + 2 * vi];
        vf.u[4 + vi] = *(const unsigned int*)&sVt[n][16 + kh + 2 * vi];
      }
      acc[c].v = __builtin_amdgcn_wmma_f32_16x16x32_bf16(false, pf.f, false, vf.f,
                                                         (short)0, acc[c].v, false, false);
    }
    __syncthreads();
  }

  const int rowb = q0 + (wave << 4) + ((lane >> 4) << 3);
#pragma unroll
  for (int r = 0; r < 8; ++r) {
    const float inv = 1.0f / lrow[r];
    const int row = rowb + r;
    if (row < qlen) {
#pragma unroll
      for (int c = 0; c < 8; ++c)
        O[(long)row * HD + colh + (c << 4) + ln] = acc[c].f[r] * inv;
    }
  }
}

// ---------------------------------------------------------------------------
// Host orchestration
// ---------------------------------------------------------------------------
extern "C" void kernel_launch(void* const* d_in, const int* in_sizes, int n_in,
                              void* d_out, int out_size, void* d_ws, size_t ws_size,
                              hipStream_t stream) {
  (void)in_sizes; (void)n_in; (void)out_size; (void)ws_size;
  constexpr long Dm = 3072, Sm = 2048, ENCm = 512, NIPm = 64, SLm = 2560, CONDm = 1024;
  constexpr long W2 = Dm * Dm;

  const float* hs  = (const float*)d_in[0];
  const float* ehs = (const float*)d_in[1];
  const float* img = (const float*)d_in[2];
  // weight order: Wq,Wk,Wv,Waq,Wak,Wav,Wo,Wao,Wkip,Wvip
  const float* Wmat[10] = {
      (const float*)d_in[3],  (const float*)d_in[5],  (const float*)d_in[7],
      (const float*)d_in[9],  (const float*)d_in[11], (const float*)d_in[13],
      (const float*)d_in[15], (const float*)d_in[17], (const float*)d_in[19],
      (const float*)d_in[20]};
  const float* bq  = (const float*)d_in[4];
  const float* bk  = (const float*)d_in[6];
  const float* bv  = (const float*)d_in[8];
  const float* baq = (const float*)d_in[10];
  const float* bak = (const float*)d_in[12];
  const float* bav = (const float*)d_in[14];
  const float* bo  = (const float*)d_in[16];
  const float* bao = (const float*)d_in[18];
  const float* lq_dn = (const float*)d_in[21];
  const float* lq_up = (const float*)d_in[22];
  const float* lk_dn = (const float*)d_in[23];
  const float* lk_up = (const float*)d_in[24];
  const float* lv_dn = (const float*)d_in[25];
  const float* lv_up = (const float*)d_in[26];
  const float* lp_dn = (const float*)d_in[27];
  const float* lp_up = (const float*)d_in[28];

  char* ws = (char*)d_ws;
  size_t off = 0;
  auto take = [&](size_t bytes) -> char* {
    char* p = ws + off;
    off = (off + bytes + 255) & ~(size_t)255;
    return p;
  };
  unsigned short* wt[10];                                   // weights, bf16, transposed [N][K]
  for (int i = 0; i < 10; ++i) wt[i] = (unsigned short*)take(W2 * 2);
  unsigned short* hs_bf  = (unsigned short*)take(Sm * Dm * 2);
  unsigned short* ehs_bf = (unsigned short*)take(ENCm * Dm * 2);
  unsigned short* img_bf = (unsigned short*)take(NIPm * Dm * 2);
  float* qcat = (float*)take(SLm * Dm * 4);
  float* kcat = (float*)take(SLm * Dm * 4);
  float* vcat = (float*)take(SLm * Dm * 4);
  float* kipf = (float*)take(NIPm * Dm * 4);
  float* vipf = (float*)take(NIPm * Dm * 4);
  unsigned short* qbf   = (unsigned short*)take(SLm * Dm * 2);
  unsigned short* kbf   = (unsigned short*)take(SLm * Dm * 2);
  unsigned short* vtbf  = (unsigned short*)take(Dm * SLm * 2);   // V^T  [3072][2560]
  unsigned short* kipbf = (unsigned short*)take(NIPm * Dm * 2);
  unsigned short* viptbf= (unsigned short*)take(Dm * NIPm * 2);  // Vip^T [3072][64]
  float* outbig = (float*)take(SLm * Dm * 4);
  float* outip  = (float*)take(Sm * Dm * 4);
  unsigned short* mainbf = (unsigned short*)take(Sm * Dm * 2);
  unsigned short* encbf  = (unsigned short*)take(ENCm * Dm * 2);
  float* lt = (float*)take(CONDm * 16 * 4);

  auto cvt = [&](const float* src, unsigned short* dst, long n) {
    cvt_bf16_kernel<<<dim3((unsigned)((n + 255) / 256)), dim3(256), 0, stream>>>(src, dst, n);
  };
  auto cvt_t = [&](const float* src, unsigned short* dst, int R, int C) {
    cvt_transpose_bf16_kernel<<<dim3((unsigned)(C / 32), (unsigned)(R / 32)), dim3(256), 0, stream>>>(
        src, dst, R, C);
  };
  for (int i = 0; i < 10; ++i) cvt_t(Wmat[i], wt[i], (int)Dm, (int)Dm);  // [K][N] -> [N][K]
  cvt(hs, hs_bf, Sm * Dm);
  cvt(ehs, ehs_bf, ENCm * Dm);
  cvt(img, img_bf, NIPm * Dm);

  auto gemm = [&](const unsigned short* A, const unsigned short* BT, float* C,
                  const float* bias, int M) {
    dim3 g((unsigned)(3072 / 64), (unsigned)((M + 127) / 128));
    gemm_bf16_kernel<<<g, dim3(256), 0, stream>>>(A, BT, C, bias, M, 3072, 3072, 3072);
  };

  // projections (enc rows 0..511, main rows 512..2559 of the concatenated buffers)
  gemm(ehs_bf, wt[3], qcat, baq, (int)ENCm);
  gemm(ehs_bf, wt[4], kcat, bak, (int)ENCm);
  gemm(ehs_bf, wt[5], vcat, bav, (int)ENCm);
  gemm(hs_bf,  wt[0], qcat + ENCm * Dm, bq, (int)Sm);
  gemm(hs_bf,  wt[1], kcat + ENCm * Dm, bk, (int)Sm);
  gemm(hs_bf,  wt[2], vcat + ENCm * Dm, bv, (int)Sm);
  gemm(img_bf, wt[8], kipf, nullptr, (int)NIPm);
  gemm(img_bf, wt[9], vipf, nullptr, (int)NIPm);

  // LoRA on last COND tokens (added before RMS), scale = LW*(ALPHA/RANK) = 1.0
  auto lora = [&](const float* X, const float* dn, const float* up, float* Y) {
    lora_dn_kernel<<<dim3((unsigned)((CONDm * 16) / 8)), dim3(256), 0, stream>>>(X, dn, lt, (int)CONDm);
    const long n = CONDm * Dm;
    lora_up_add_kernel<<<dim3((unsigned)((n + 255) / 256)), dim3(256), 0, stream>>>(lt, up, Y, n, 1.0f);
  };
  const float* hs_tail = hs + (Sm - CONDm) * Dm;
  lora(hs_tail, lq_dn, lq_up, qcat + (SLm - CONDm) * Dm);
  lora(hs_tail, lk_dn, lk_up, kcat + (SLm - CONDm) * Dm);
  lora(hs_tail, lv_dn, lv_up, vcat + (SLm - CONDm) * Dm);

  // per-head RMS (q, k full concat; kip)
  rms_kernel<<<dim3((unsigned)((SLm * 24) / 8)), dim3(256), 0, stream>>>(qcat, SLm * 24);
  rms_kernel<<<dim3((unsigned)((SLm * 24) / 8)), dim3(256), 0, stream>>>(kcat, SLm * 24);
  rms_kernel<<<dim3((unsigned)((NIPm * 24) / 8)), dim3(256), 0, stream>>>(kipf, NIPm * 24);

  cvt(qcat, qbf, SLm * Dm);
  cvt(kcat, kbf, SLm * Dm);
  cvt(kipf, kipbf, NIPm * Dm);
  cvt_t(vcat, vtbf, (int)SLm, (int)Dm);    // [2560][3072] -> [3072][2560]
  cvt_t(vipf, viptbf, (int)NIPm, (int)Dm); // [64][3072]   -> [3072][64]

  const float scale = 0.08838834764831845f;  // 1/sqrt(128)
  // big masked attention: rows >= 1536 see only cols >= 1536 (tile-aligned)
  attn_kernel<<<dim3((unsigned)(SLm / 128), 24), dim3(256), 0, stream>>>(
      qbf, kbf, vtbf, outbig, (int)SLm, (int)SLm, 1536, 1536, scale);
  // IP attention: main q rows vs 64 image tokens, unmasked
  attn_kernel<<<dim3((unsigned)(Sm / 128), 24), dim3(256), 0, stream>>>(
      qbf + ENCm * Dm, kipbf, viptbf, outip, (int)Sm, (int)NIPm, 1 << 30, 0, scale);

  // main = attn_main + IP_SCALE * ip_out  (IP_SCALE = 1.0) -> bf16
  add_cvt_bf16_kernel<<<dim3((unsigned)((Sm * Dm + 255) / 256)), dim3(256), 0, stream>>>(
      outbig + ENCm * Dm, outip, mainbf, Sm * Dm);

  float* out_f = (float*)d_out;
  // main projection straight into d_out rows 512..2559
  gemm(mainbf, wt[6], out_f + ENCm * Dm, bo, (int)Sm);
  // post-projection LoRA on last COND rows of main (in place on d_out)
  lora(out_f + (ENCm + Sm - CONDm) * Dm, lp_dn, lp_up, out_f + (ENCm + Sm - CONDm) * Dm);

  // encoder output projection into d_out rows 0..511
  cvt(outbig, encbf, ENCm * Dm);
  gemm(encbf, wt[7], out_f, bao, (int)ENCm);
}
